// SparseMoE_45715631898871
// MI455X (gfx1250) — compile-verified, matrix operand order checked by reading, not run on hardware
//
#include <hip/hip_runtime.h>
#include <math.h>

// ---------------- problem constants (from reference) ----------------
#define T_TOK 8192          // B*S
#define DIM   1024          // D
#define NEXP  8             // E
#define HDIM  4096          // H
#define CAP   2252          // int(T*K/E*1.1)
#define CAPP  2304          // CAP rounded up to multiple of BM (36 tiles of 64)
#define OUT_N (T_TOK * DIM) // flat output elements before the scalar loss

// GEMM tiling
#define BM    64            // block M tile (capacity slots)
#define BN    256           // block N tile
#define KC    64            // K chunk staged in LDS
#define LDK   (KC + 8)      // padded LDS row stride: 144B -> 16B aligned, bank-spread

typedef __attribute__((ext_vector_type(16))) __bf16 v16bf;
typedef __attribute__((ext_vector_type(8)))  float  v8f;

// ---------------- zero output + loss sums ----------------
__global__ void k_zero(float* __restrict__ out, float* __restrict__ sums, int n) {
    int stride = gridDim.x * blockDim.x;
    for (int i = blockIdx.x * blockDim.x + threadIdx.x; i < n; i += stride)
        out[i] = 0.f;
    int g = blockIdx.x * blockDim.x + threadIdx.x;
    if (g < 16) sums[g] = 0.f;
}

// ---------------- router: wave32 per token ----------------
__global__ void __launch_bounds__(256)
k_router(const float* __restrict__ x, const float* __restrict__ noise,
         const float* __restrict__ Wr, const float* __restrict__ br,
         const float* __restrict__ Wn, const float* __restrict__ bn,
         int* __restrict__ top_idx, float* __restrict__ top_gate,
         float* __restrict__ sum_rp, float* __restrict__ sum_mask) {
    int wave = threadIdx.x >> 5;
    int lane = threadIdx.x & 31;
    int t = blockIdx.x * 8 + wave;
    const float* xr = x + (size_t)t * DIM;

    float ar[NEXP], an[NEXP];
#pragma unroll
    for (int e = 0; e < NEXP; e++) { ar[e] = 0.f; an[e] = 0.f; }

    for (int i = lane; i < DIM; i += 32) {
        float xv = xr[i];
        const float* wr = Wr + i * NEXP;
        const float* wn = Wn + i * NEXP;
#pragma unroll
        for (int e = 0; e < NEXP; e++) {
            ar[e] += xv * wr[e];
            an[e] += xv * wn[e];
        }
    }
#pragma unroll
    for (int off = 16; off > 0; off >>= 1) {
#pragma unroll
        for (int e = 0; e < NEXP; e++) {
            ar[e] += __shfl_down(ar[e], off, 32);
            an[e] += __shfl_down(an[e], off, 32);
        }
    }
    if (lane == 0) {
        float noisy[NEXP];
#pragma unroll
        for (int e = 0; e < NEXP; e++) {
            float nl = an[e] + bn[e];
            float sp = fmaxf(nl, 0.f) + log1pf(expf(-fabsf(nl)));  // stable softplus
            noisy[e] = ar[e] + br[e] + noise[(size_t)t * NEXP + e] * sp;
        }
        // top-2 with jax tie-break (lowest index wins on ties)
        int i1 = 0; float v1 = noisy[0];
#pragma unroll
        for (int e = 1; e < NEXP; e++) if (noisy[e] > v1) { v1 = noisy[e]; i1 = e; }
        int i2 = (i1 == 0) ? 1 : 0; float v2 = noisy[i2];
#pragma unroll
        for (int e = 0; e < NEXP; e++)
            if (e != i1 && e != i2 && noisy[e] > v2) { v2 = noisy[e]; i2 = e; }
        float p1 = 1.f / (1.f + expf(v2 - v1));   // softmax over {v1, v2}, v1 >= v2
        float p2 = 1.f - p1;
        top_idx[t * 2 + 0] = i1;  top_idx[t * 2 + 1] = i2;
        top_gate[t * 2 + 0] = p1; top_gate[t * 2 + 1] = p2;
        atomicAdd(&sum_rp[i1], p1);    atomicAdd(&sum_rp[i2], p2);
        atomicAdd(&sum_mask[i1], 1.f); atomicAdd(&sum_mask[i2], 1.f);
    }
}

// ---------------- capacity assignment: one wave per expert, ballot scan ----------------
__global__ void k_assign(const int* __restrict__ top_idx, const float* __restrict__ top_gate,
                         int* __restrict__ gather_idx, float* __restrict__ gate_sel,
                         int* __restrict__ counts) {
    int e = blockIdx.x;
    int lane = threadIdx.x;   // 32 threads
    int base = 0;
    for (int c = 0; c < T_TOK / 32; c++) {
        int t = c * 32 + lane;
        int i1 = top_idx[t * 2 + 0];
        int i2 = top_idx[t * 2 + 1];
        bool m1 = (i1 == e);
        bool m2 = (i2 == e);
        bool m = m1 || m2;
        unsigned mask = (unsigned)__ballot(m);
        int prefix = __popc(mask & ((1u << lane) - 1u));
        int slot = base + prefix;
        if (m && slot < CAP) {
            gather_idx[e * CAPP + slot] = t;
            gate_sel[e * CAPP + slot] = m1 ? top_gate[t * 2] : top_gate[t * 2 + 1];
        }
        base += __popc(mask);
    }
    int cnt = base < CAP ? base : CAP;
    for (int s = cnt + lane; s < CAPP; s += 32) {  // padded / overflow slots: inert
        gather_idx[e * CAPP + s] = 0;
        gate_sel[e * CAPP + s] = 0.f;
    }
    if (lane == 0) counts[e] = cnt;
}

// ---------------- load-balance loss ----------------
__global__ void k_lb(const float* __restrict__ sums, float* __restrict__ out) {
    if (threadIdx.x == 0 && blockIdx.x == 0) {
        float lb = 0.f;
#pragma unroll
        for (int e = 0; e < NEXP; e++)
            lb += (sums[e] / (float)T_TOK) * (sums[8 + e] / (float)T_TOK);
        out[OUT_N] = (float)NEXP * lb;
    }
}

// ---------------- WMMA fragment helper (CDNA5 16x16x32 bf16 layout) ----------------
// A (16x32): lane<16 -> row=lane,   vec[0..7]=K kh..kh+7, vec[8..15]=K 16+kh..23+kh,
//            kh=(lane>>4)*8. B (32x16) mirrors with column = lane&15.
__device__ __forceinline__ v16bf frag_from_row(const __bf16* row, int kbase, int kh) {
    v16bf f;
#pragma unroll
    for (int i = 0; i < 8; i++) {
        f[i]     = row[kbase + kh + i];
        f[8 + i] = row[kbase + 16 + kh + i];
    }
    return f;
}

// ---------------- GEMM1: h = relu(gather(x) @ W1[e] + b1[e]) -> bf16 ----------------
// block = 256 thr / 8 waves; block tile 64x256; wave tile 32x64 (2x4 wmma tiles)
__global__ void __launch_bounds__(256)
k_ffn1(int e, const float* __restrict__ x, const float* __restrict__ W1,
       const float* __restrict__ b1, const int* __restrict__ gather_idx,
       __bf16* __restrict__ hbuf) {
    __shared__ __bf16 As[BM][LDK];    // token rows (bf16)
    __shared__ __bf16 Bt[BN][LDK];    // W1 chunk transposed: [n][k]
    __shared__ int toks[BM];

    int tid  = threadIdx.x;
    int wave = tid >> 5, lane = tid & 31;
    int wr = wave >> 2, wc = wave & 3;           // wave grid 2x4 over 64x256
    int n_base = blockIdx.x * BN;                // over H
    int m0 = blockIdx.y * BM;                    // over capacity slots
    if (tid < BM) toks[tid] = gather_idx[e * CAPP + m0 + tid];
    __syncthreads();

    const float* W = W1 + (size_t)e * DIM * HDIM;
    v8f acc[2][4] = {};

    int kh = (lane >> 4) * 8;
    int nl = lane & 15;

    for (int k0 = 0; k0 < DIM; k0 += KC) {
        // stage A: BM x KC fp32 (gathered token rows) -> bf16 LDS
#pragma unroll
        for (int i = 0; i < (BM * KC) / 256; i++) {          // 16 elems/thread
            int idx = tid + 256 * i;
            int r = idx >> 6, c = idx & 63;
            As[r][c] = (__bf16)x[(size_t)toks[r] * DIM + k0 + c];
        }
        // stage B: KC x BN fp32 -> bf16, transposed into Bt[n][k]
#pragma unroll 4
        for (int i = 0; i < (KC * BN) / 256; i++) {          // 64 elems/thread
            int idx = tid + 256 * i;
            int kr = idx >> 8, cn = idx & 255;
            Bt[cn][kr] = (__bf16)W[(size_t)(k0 + kr) * HDIM + n_base + cn];
        }
        if (k0 + KC < DIM)   // prefetch next weight chunk (global_prefetch_b8)
            __builtin_prefetch(&W[(size_t)(k0 + KC) * HDIM + n_base + tid]);
        __syncthreads();

#pragma unroll
        for (int kk = 0; kk < KC / 32; kk++) {
            v16bf a[2], b[4];
#pragma unroll
            for (int mi = 0; mi < 2; mi++)
                a[mi] = frag_from_row(&As[wr * 32 + mi * 16 + nl][0], kk * 32, kh);
#pragma unroll
            for (int ni = 0; ni < 4; ni++)
                b[ni] = frag_from_row(&Bt[wc * 64 + ni * 16 + nl][0], kk * 32, kh);
#pragma unroll
            for (int mi = 0; mi < 2; mi++)
#pragma unroll
                for (int ni = 0; ni < 4; ni++)
                    acc[mi][ni] = __builtin_amdgcn_wmma_f32_16x16x32_bf16(
                        false, a[mi], false, b[ni], (short)0, acc[mi][ni], false, false);
        }
        __syncthreads();
    }

    // epilogue: +bias, relu, bf16 store. C layout: vgpr r -> M = r + 8*(lane>=16), N = lane&15
    int mhi = (lane >> 4) * 8;
#pragma unroll
    for (int ni = 0; ni < 4; ni++) {
        int ncol = n_base + wc * 64 + ni * 16 + nl;
        float bias = b1[(size_t)e * HDIM + ncol];
#pragma unroll
        for (int mi = 0; mi < 2; mi++) {
#pragma unroll
            for (int r = 0; r < 8; r++) {
                int m = m0 + wr * 32 + mi * 16 + r + mhi;
                float v = acc[mi][ni][r] + bias;
                hbuf[(size_t)m * HDIM + ncol] = (__bf16)fmaxf(v, 0.f);
            }
        }
    }
}

// ---------------- GEMM2: out += gate * (h @ W2[e] + b2[e]) via atomic scatter ----------------
__global__ void __launch_bounds__(256)
k_ffn2(int e, const __bf16* __restrict__ hbuf, const float* __restrict__ W2,
       const float* __restrict__ b2, const int* __restrict__ gather_idx,
       const float* __restrict__ gate_sel, const int* __restrict__ counts,
       float* __restrict__ out) {
    __shared__ __bf16 As[BM][LDK];
    __shared__ __bf16 Bt[BN][LDK];

    int tid  = threadIdx.x;
    int wave = tid >> 5, lane = tid & 31;
    int wr = wave >> 2, wc = wave & 3;
    int n_base = blockIdx.x * BN;                // over D
    int m0 = blockIdx.y * BM;

    const float* W = W2 + (size_t)e * HDIM * DIM;
    v8f acc[2][4] = {};

    int kh = (lane >> 4) * 8;
    int nl = lane & 15;

    for (int k0 = 0; k0 < HDIM; k0 += KC) {
#pragma unroll
        for (int i = 0; i < (BM * KC) / 256; i++) {
            int idx = tid + 256 * i;
            int r = idx >> 6, c = idx & 63;
            As[r][c] = hbuf[(size_t)(m0 + r) * HDIM + k0 + c];
        }
#pragma unroll 4
        for (int i = 0; i < (KC * BN) / 256; i++) {
            int idx = tid + 256 * i;
            int kr = idx >> 8, cn = idx & 255;
            Bt[cn][kr] = (__bf16)W[(size_t)(k0 + kr) * DIM + n_base + cn];
        }
        if (k0 + KC < HDIM)
            __builtin_prefetch(&W[(size_t)(k0 + KC) * DIM + n_base + tid]);
        __syncthreads();

#pragma unroll
        for (int kk = 0; kk < KC / 32; kk++) {
            v16bf a[2], b[4];
#pragma unroll
            for (int mi = 0; mi < 2; mi++)
                a[mi] = frag_from_row(&As[wr * 32 + mi * 16 + nl][0], kk * 32, kh);
#pragma unroll
            for (int ni = 0; ni < 4; ni++)
                b[ni] = frag_from_row(&Bt[wc * 64 + ni * 16 + nl][0], kk * 32, kh);
#pragma unroll
            for (int mi = 0; mi < 2; mi++)
#pragma unroll
                for (int ni = 0; ni < 4; ni++)
                    acc[mi][ni] = __builtin_amdgcn_wmma_f32_16x16x32_bf16(
                        false, a[mi], false, b[ni], (short)0, acc[mi][ni], false, false);
        }
        __syncthreads();
    }

    int cnt = counts[e];
    int mhi = (lane >> 4) * 8;
    float bias[4];
#pragma unroll
    for (int ni = 0; ni < 4; ni++)
        bias[ni] = b2[(size_t)e * DIM + n_base + wc * 64 + ni * 16 + nl];

#pragma unroll
    for (int mi = 0; mi < 2; mi++) {
#pragma unroll
        for (int r = 0; r < 8; r++) {
            int m = m0 + wr * 32 + mi * 16 + r + mhi;
            if (m < cnt) {
                int tok = gather_idx[e * CAPP + m];
                float g = gate_sel[e * CAPP + m];
#pragma unroll
                for (int ni = 0; ni < 4; ni++) {
                    int ncol = n_base + wc * 64 + ni * 16 + nl;
                    atomicAdd(&out[(size_t)tok * DIM + ncol],
                              (acc[mi][ni][r] + bias[ni]) * g);
                }
            }
        }
    }
}

// ---------------- host-side orchestration ----------------
extern "C" void kernel_launch(void* const* d_in, const int* in_sizes, int n_in,
                              void* d_out, int out_size, void* d_ws, size_t ws_size,
                              hipStream_t stream) {
    (void)in_sizes; (void)n_in; (void)out_size; (void)ws_size;
    const float* x     = (const float*)d_in[0];
    const float* noise = (const float*)d_in[1];
    const float* Wr    = (const float*)d_in[2];
    const float* br    = (const float*)d_in[3];
    const float* Wn    = (const float*)d_in[4];
    const float* bn    = (const float*)d_in[5];
    const float* W1    = (const float*)d_in[6];
    const float* b1    = (const float*)d_in[7];
    const float* W2    = (const float*)d_in[8];
    const float* b2    = (const float*)d_in[9];
    float* out = (float*)d_out;

    // workspace layout
    char* ws = (char*)d_ws;
    float* sums       = (float*)ws;                        // [16]: rp sums then mask sums
    int*   counts     = (int*)(ws + 64);                   // [8]
    int*   top_idx    = (int*)(ws + 96);                   // [T*2]
    float* top_gate   = (float*)(ws + 96 + T_TOK * 2 * 4); // [T*2]
    size_t off = 96 + (size_t)T_TOK * 4 * 4;
    int*   gather_idx = (int*)(ws + off);                  // [E*CAPP]
    off += (size_t)NEXP * CAPP * 4;
    float* gate_sel   = (float*)(ws + off);                // [E*CAPP]
    off += (size_t)NEXP * CAPP * 4;
    off = (off + 255) & ~(size_t)255;
    __bf16* hbuf      = (__bf16*)(ws + off);               // [CAPP*H] bf16, reused per expert

    k_zero<<<4096, 256, 0, stream>>>(out, sums, OUT_N + 1);
    k_router<<<T_TOK / 8, 256, 0, stream>>>(x, noise, Wr, br, Wn, bn,
                                            top_idx, top_gate, sums, sums + 8);
    k_assign<<<NEXP, 32, 0, stream>>>(top_idx, top_gate, gather_idx, gate_sel, counts);
    k_lb<<<1, 1, 0, stream>>>(sums, out);

    dim3 g1(HDIM / BN, CAPP / BM);   // 16 x 36
    dim3 g2(DIM  / BN, CAPP / BM);   //  4 x 36
    for (int e = 0; e < NEXP; e++) {
        k_ffn1<<<g1, 256, 0, stream>>>(e, x, W1, b1, gather_idx, hbuf);
        k_ffn2<<<g2, 256, 0, stream>>>(e, hbuf, W2, b2, gather_idx, gate_sel, counts, out);
    }
}
